// DehomoCodingGenerator_58789512347730
// MI455X (gfx1250) — matrix operand order repaired and checked by reading, not compiled, not run on hardware
//
#include <hip/hip_runtime.h>
#include <float.h>

#define D      256
#define NQc    1000
#define BSc    8
#define TOPK   10
#define THR    0.5f

typedef __attribute__((ext_vector_type(16))) _Float16 v16h;
typedef __attribute__((ext_vector_type(8)))  _Float16 v8h;
typedef __attribute__((ext_vector_type(8)))  float    v8f;

union AFrag { v16h v; v8h h8[2]; _Float16 h[16]; };

// ---- CDNA5 async global->LDS copy (ASYNCcnt path), 16 bytes per op ----------
static __device__ inline void async_copy16(unsigned int ldsOff, const void* gaddr) {
  asm volatile("global_load_async_to_lds_b128 %0, %1, off"
               :: "v"(ldsOff), "v"(gaddr) : "memory");
}
static __device__ inline void wait_async0() {
  asm volatile("s_wait_asynccnt 0x0" ::: "memory");
}
static __device__ inline unsigned int lds_off(const void* p) {
  return (unsigned int)(unsigned long long)p;  // LDS aperture: low 32 bits = LDS byte offset
}

// A fragment: 16x32 f16, row-major source with leading dim ld (halfs).
// lane<16: M=lane, K = kBase + {0..7, 16..23}; lane>=16: K = kBase + {8..15, 24..31}
static __device__ inline v16h ldA_f16(const _Float16* tile, int ld, int kBase, int lane) {
  int g = (lane >> 4) & 1, m = lane & 15;
  const _Float16* row = tile + m * ld + kBase + g * 8;
  AFrag a;
  a.h8[0] = *(const v8h*)(row);
  a.h8[1] = *(const v8h*)(row + 16);
  return a.v;
}

static __device__ inline v16h ldA_f32(const float* tile, int ld, int kBase, int lane) {
  int g = (lane >> 4) & 1, m = lane & 15;
  const float* row = tile + m * ld + kBase + g * 8;
  AFrag a;
#pragma unroll
  for (int t = 0; t < 8; t++) {
    a.h[t]     = (_Float16)row[t];
    a.h[8 + t] = (_Float16)row[16 + t];
  }
  return a.v;
}

// B fragment: 32x16 f16 from transposed weights Wt[N][K] (row-major, ld halfs).
// lane%16 = N; lane<16: K = kBase + {0..7, 16..23}; lane>=16: K = kBase + {8..15, 24..31}
static __device__ inline v16h ldB(const _Float16* wtTile, int ld, int kBase, int lane) {
  int g = (lane >> 4) & 1, n = lane & 15;
  const _Float16* p = wtTile + n * ld + kBase + g * 8;
  AFrag b;
  b.h8[0] = *(const v8h*)(p);
  b.h8[1] = *(const v8h*)(p + 16);
  return b.v;
}

static __device__ inline v8f mm_f16(const _Float16* A, int lda, const _Float16* Bt, int ldb, int lane) {
  __builtin_prefetch(Bt, 0, 3);
  v8f c;
#pragma unroll
  for (int r = 0; r < 8; r++) c[r] = 0.0f;
#pragma unroll
  for (int k = 0; k < D; k += 32) {
    v16h a = ldA_f16(A, lda, k, lane);
    v16h b = ldB(Bt, ldb, k, lane);
    c = __builtin_amdgcn_wmma_f32_16x16x32_f16(false, a, false, b, (short)0, c, false, false);
  }
  return c;
}

static __device__ inline v8f mm_f32A(const float* A, int lda, const _Float16* Bt, int ldb, int lane) {
  __builtin_prefetch(Bt, 0, 3);
  v8f c;
#pragma unroll
  for (int r = 0; r < 8; r++) c[r] = 0.0f;
#pragma unroll
  for (int k = 0; k < D; k += 32) {
    v16h a = ldA_f32(A, lda, k, lane);
    v16h b = ldB(Bt, ldb, k, lane);
    c = __builtin_amdgcn_wmma_f32_16x16x32_f16(false, a, false, b, (short)0, c, false, false);
  }
  return c;
}

// ---------------- weight convert + transpose: Wt[n*256+k] = (f16)W[k*256+n] -----
__global__ void k_wconv(const float* W1, const float* W2, const float* W3,
                        const float* W4, const float* W5, _Float16* wt) {
  int id = blockIdx.x * blockDim.x + threadIdx.x;  // 5 * 65536
  int w = id >> 16, rem = id & 65535;
  int k = rem >> 8, n = rem & 255;
  const float* W = (w == 0) ? W1 : (w == 1) ? W2 : (w == 2) ? W3 : (w == 3) ? W4 : W5;
  wt[(size_t)w * 65536 + n * 256 + k] = (_Float16)W[k * 256 + n];
}

// ---------------- init max accumulator ---------------------------------------
__global__ void k_init(float* p, int n) {
  int i = blockIdx.x * blockDim.x + threadIdx.x;
  if (i < n) p[i] = -3.402823466e38f;
}

// ---------------- IoU + attn_mask + stable top-10 ----------------------------
__global__ void k_topk(const float* boxes, const float* seed, float* maskOut,
                       int* idxb, float* mkv) {
  __shared__ float sv[32 * TOPK];
  __shared__ int   si[32 * TOPK];
  int lane = threadIdx.x;          // block = 32
  int row  = blockIdx.x;           // 0..7999
  int b = row / NQc;
  const float* bi = boxes + (size_t)row * 4;
  float x0 = bi[0] - 0.5f * bi[2], y0 = bi[1] - 0.5f * bi[3];
  float x1 = bi[0] + 0.5f * bi[2], y1 = bi[1] + 0.5f * bi[3];
  float ai = (x1 - x0) * (y1 - y0);
  float negi = 1.0f - seed[row];

  float tv[TOPK]; int ti[TOPK];
#pragma unroll
  for (int t = 0; t < TOPK; t++) { tv[t] = -1.0f; ti[t] = 0x7fffffff; }

  for (int j = lane; j < NQc; j += 32) {
    const float* bj = boxes + (size_t)(b * NQc + j) * 4;
    float jx0 = bj[0] - 0.5f * bj[2], jy0 = bj[1] - 0.5f * bj[3];
    float jx1 = bj[0] + 0.5f * bj[2], jy1 = bj[1] + 0.5f * bj[3];
    float aj = (jx1 - jx0) * (jy1 - jy0);
    float lx = fmaxf(x0, jx0), ly = fmaxf(y0, jy0);
    float rx = fminf(x1, jx1), ry = fminf(y1, jy1);
    float iw = fmaxf(rx - lx, 0.0f), ih = fmaxf(ry - ly, 0.0f);
    float inter = iw * ih;
    float iou = inter / (ai + aj - inter);
    maskOut[(size_t)row * NQc + j] = (iou >= THR) ? 1.0f : 0.0f;
    float ov = iou * seed[b * NQc + j] * negi;
    if (ov > tv[TOPK - 1] || (ov == tv[TOPK - 1] && j < ti[TOPK - 1])) {
      tv[TOPK - 1] = ov; ti[TOPK - 1] = j;
#pragma unroll
      for (int s = TOPK - 1; s > 0; s--) {
        bool bet = tv[s] > tv[s - 1] || (tv[s] == tv[s - 1] && ti[s] < ti[s - 1]);
        if (bet) {
          float fv = tv[s]; tv[s] = tv[s - 1]; tv[s - 1] = fv;
          int   fi = ti[s]; ti[s] = ti[s - 1]; ti[s - 1] = fi;
        }
      }
    }
  }
#pragma unroll
  for (int t = 0; t < TOPK; t++) { sv[lane * TOPK + t] = tv[t]; si[lane * TOPK + t] = ti[t]; }
  __syncthreads();

  for (int t = 0; t < TOPK; t++) {
    float bv = -2.0f; int bj = 0x7fffffff; int bs = -1;
    for (int s = lane; s < 32 * TOPK; s += 32) {
      float v = sv[s]; int jj = si[s];
      if (v > bv || (v == bv && jj < bj)) { bv = v; bj = jj; bs = s; }
    }
#pragma unroll
    for (int off = 16; off > 0; off >>= 1) {
      float ov2 = __shfl_down(bv, off);
      int   oj  = __shfl_down(bj, off);
      int   os  = __shfl_down(bs, off);
      if (ov2 > bv || (ov2 == bv && oj < bj)) { bv = ov2; bj = oj; bs = os; }
    }
    if (lane == 0) {
      idxb[row * TOPK + t] = bj;
      float sj = seed[b * NQc + bj];
      mkv[row * TOPK + t] = sj * ((bv >= THR) ? 1.0f : 0.0f);
      sv[bs] = -2.0f; si[bs] = 0x7fffffff;
    }
    __syncthreads();
  }
}

// ---------------- H1 = relu(tgt@W1 + b1), f16 out; A staged via async->LDS ----
__global__ void k_gemm1(const float* tgt, const _Float16* w1t, const float* b1, _Float16* h1) {
  __shared__ __align__(16) float As[16 * D];   // 16KB f32 A tile, shared by 8 waves
  int lane = threadIdx.x & 31, wave = threadIdx.x >> 5;
  int tM = blockIdx.x >> 1;
  int tN = (blockIdx.x & 1) * 8 + wave;
  {
    const char* g = (const char*)(tgt + (size_t)tM * 16 * D);
    unsigned int l0 = lds_off(As);
#pragma unroll
    for (int t = threadIdx.x; t < 1024; t += 256)   // 16KB / 16B
      async_copy16(l0 + t * 16, g + t * 16);
    wait_async0();
  }
  __syncthreads();
  v8f c = mm_f32A(As, D, w1t + (size_t)tN * 16 * D, D, lane);
  int g = (lane >> 4) & 1, n = lane & 15;
  int nG = tN * 16 + n;
  float bias = b1[nG];
#pragma unroll
  for (int r = 0; r < 8; r++) {
    int mG = tM * 16 + g * 8 + r;
    float v = c[r] + bias; v = v > 0.0f ? v : 0.0f;
    h1[(size_t)mG * D + nG] = (_Float16)v;
  }
}

// ---------------- T = H1@W2 + b2, f32 out; A staged via async->LDS ------------
__global__ void k_gemm2(const _Float16* h1, const _Float16* w2t, const float* b2, float* T) {
  __shared__ __align__(16) _Float16 As[16 * D];  // 8KB f16 A tile
  int lane = threadIdx.x & 31, wave = threadIdx.x >> 5;
  int tM = blockIdx.x >> 1;
  int tN = (blockIdx.x & 1) * 8 + wave;
  {
    const char* g = (const char*)(h1 + (size_t)tM * 16 * D);
    unsigned int l0 = lds_off(As);
#pragma unroll
    for (int t = threadIdx.x; t < 512; t += 256)    // 8KB / 16B
      async_copy16(l0 + t * 16, g + t * 16);
    wait_async0();
  }
  __syncthreads();
  v8f c = mm_f16(As, D, w2t + (size_t)tN * 16 * D, D, lane);
  int g = (lane >> 4) & 1, n = lane & 15;
  int nG = tN * 16 + n;
  float bias = b2[nG];
#pragma unroll
  for (int r = 0; r < 8; r++) {
    int mG = tM * 16 + g * 8 + r;
    T[(size_t)mG * D + nG] = c[r] + bias;
  }
}

// ---------------- id_token = LayerNorm(T) ------------------------------------
__global__ void k_ln(const float* T, const float* g2, const float* be2, float* idt) {
  int lane = threadIdx.x & 31, wave = threadIdx.x >> 5;
  int row = blockIdx.x * 8 + wave;
  const float* x = T + (size_t)row * D;
  float v[8], s = 0.0f, s2 = 0.0f;
#pragma unroll
  for (int t = 0; t < 8; t++) { v[t] = x[lane * 8 + t]; s += v[t]; s2 += v[t] * v[t]; }
#pragma unroll
  for (int off = 16; off > 0; off >>= 1) { s += __shfl_xor(s, off); s2 += __shfl_xor(s2, off); }
  float mean = s * (1.0f / D);
  float var  = s2 * (1.0f / D) - mean * mean;
  float rs   = rsqrtf(var + 1e-5f);
#pragma unroll
  for (int t = 0; t < 8; t++) {
    int cI = lane * 8 + t;
    idt[(size_t)row * D + cI] = (v[t] - mean) * rs * g2[cI] + be2[cI];
  }
}

static __device__ inline void atomMaxF(float* addr, float val) { atomicMax(addr, val); }

// ---------------- fused: diff -> relu(@W3+b3) -> @W4+b4 -> mk/max ------------
__global__ void k_feature(const float* idt, const _Float16* w3t, const float* b3,
                          const _Float16* w4t, const float* b4,
                          const int* idxb, const float* mkv, float* curmax) {
  __shared__ __align__(16) _Float16 Dst[16 * D];
  __shared__ __align__(16) _Float16 F1[16 * D];
  int lane = threadIdx.x & 31, wave = threadIdx.x >> 5;
  int strip = blockIdx.x;  // 0..4999, each = 16 rows of the 80000-row diff matrix

  // stage A: gather diffs into LDS
  {
    int row = threadIdx.x >> 4;          // 0..15
    int c0  = (threadIdx.x & 15) * 16;   // 16 consecutive cols
    int rg = strip * 16 + row;
    int q = rg / TOPK, k = rg % TOPK;
    int b = q / NQc;
    int j = idxb[q * TOPK + k];
    const float* xi = idt + (size_t)q * D;
    const float* xj = idt + (size_t)(b * NQc + j) * D;
#pragma unroll
    for (int t = 0; t < 16; t++)
      Dst[row * D + c0 + t] = (_Float16)(xi[c0 + t] - xj[c0 + t]);
  }
  __syncthreads();

  // stage B: F1 = relu(Dst @ W3 + b3) -> LDS (f16)
#pragma unroll
  for (int tt = 0; tt < 2; tt++) {
    int tN = wave + tt * 8;
    v8f c = mm_f16(Dst, D, w3t + (size_t)tN * 16 * D, D, lane);
    int g = (lane >> 4) & 1, n = lane & 15;
    int nG = tN * 16 + n;
    float bias = b3[nG];
#pragma unroll
    for (int r = 0; r < 8; r++) {
      float v = c[r] + bias; v = v > 0.0f ? v : 0.0f;
      F1[(g * 8 + r) * D + nG] = (_Float16)v;
    }
  }
  __syncthreads();

  // stage C: F2 = F1 @ W4 + b4; apply mk and atomic-max into per-query accumulator
#pragma unroll
  for (int tt = 0; tt < 2; tt++) {
    int tN = wave + tt * 8;
    v8f c = mm_f16(F1, D, w4t + (size_t)tN * 16 * D, D, lane);
    int g = (lane >> 4) & 1, n = lane & 15;
    int nG = tN * 16 + n;
    float bias = b4[nG];
#pragma unroll
    for (int r = 0; r < 8; r++) {
      int rg = strip * 16 + g * 8 + r;
      int q = rg / TOPK;
      float v = (c[r] + bias) * mkv[rg];
      atomMaxF(&curmax[(size_t)q * D + nG], v);
    }
  }
}

// ---------------- out = tgt + relu(curmax@W5 + b5) * (1 - seed) --------------
__global__ void k_final(const float* curmax, const _Float16* w5t, const float* b5,
                        const float* tgt, const float* seed, float* out) {
  __shared__ __align__(16) float As[16 * D];   // 16KB f32 A tile
  int lane = threadIdx.x & 31, wave = threadIdx.x >> 5;
  int tM = blockIdx.x >> 1;
  int tN = (blockIdx.x & 1) * 8 + wave;
  {
    const char* g = (const char*)(curmax + (size_t)tM * 16 * D);
    unsigned int l0 = lds_off(As);
#pragma unroll
    for (int t = threadIdx.x; t < 1024; t += 256)
      async_copy16(l0 + t * 16, g + t * 16);
    wait_async0();
  }
  __syncthreads();
  v8f c = mm_f32A(As, D, w5t + (size_t)tN * 16 * D, D, lane);
  int g = (lane >> 4) & 1, n = lane & 15;
  int nG = tN * 16 + n;
  float bias = b5[nG];
#pragma unroll
  for (int r = 0; r < 8; r++) {
    int mG = tM * 16 + g * 8 + r;
    float v = c[r] + bias; v = v > 0.0f ? v : 0.0f;
    out[(size_t)mG * D + nG] = tgt[(size_t)mG * D + nG] + v * (1.0f - seed[mG]);
  }
}

extern "C" void kernel_launch(void* const* d_in, const int* in_sizes, int n_in,
                              void* d_out, int out_size, void* d_ws, size_t ws_size,
                              hipStream_t stream) {
  const float* tgt   = (const float*)d_in[0];
  const float* seed  = (const float*)d_in[1];
  const float* boxes = (const float*)d_in[2];
  const float* W1 = (const float*)d_in[3],  *b1 = (const float*)d_in[4];
  const float* W2 = (const float*)d_in[5],  *b2 = (const float*)d_in[6];
  const float* g2 = (const float*)d_in[7],  *be2 = (const float*)d_in[8];
  const float* W3 = (const float*)d_in[9],  *b3 = (const float*)d_in[10];
  const float* W4 = (const float*)d_in[11], *b4 = (const float*)d_in[12];
  const float* W5 = (const float*)d_in[13], *b5 = (const float*)d_in[14];

  float* outTgt  = (float*)d_out;
  float* outMask = outTgt + (size_t)BSc * NQc * D;

  const int NROW = BSc * NQc;          // 8000
  char* ws = (char*)d_ws;
  size_t off = 0;
  auto take = [&](size_t bytes) -> void* {
    void* p = ws + off;
    off = (off + bytes + 255) & ~(size_t)255;
    return p;
  };
  _Float16* wt   = (_Float16*)take((size_t)5 * 65536 * sizeof(_Float16));
  int*      idxb = (int*)     take((size_t)NROW * TOPK * sizeof(int));
  float*    mkv  = (float*)   take((size_t)NROW * TOPK * sizeof(float));
  _Float16* h1   = (_Float16*)take((size_t)NROW * D * sizeof(_Float16));
  float*    Tbuf = (float*)   take((size_t)NROW * D * sizeof(float));
  float*    idt  = (float*)   take((size_t)NROW * D * sizeof(float));
  float*    curmax = Tbuf;    // T consumed by LN before curmax is needed

  _Float16* w1t = wt;
  _Float16* w2t = wt + 1 * 65536;
  _Float16* w3t = wt + 2 * 65536;
  _Float16* w4t = wt + 3 * 65536;
  _Float16* w5t = wt + 4 * 65536;

  k_wconv<<<1280, 256, 0, stream>>>(W1, W2, W3, W4, W5, wt);
  k_topk<<<NROW, 32, 0, stream>>>(boxes, seed, outMask, idxb, mkv);
  k_gemm1<<<NROW / 16 * 2, 256, 0, stream>>>(tgt, w1t, b1, h1);         // 1000 blocks
  k_gemm2<<<NROW / 16 * 2, 256, 0, stream>>>(h1, w2t, b2, Tbuf);
  k_ln<<<NROW / 8, 256, 0, stream>>>(Tbuf, g2, be2, idt);
  k_init<<<(NROW * D + 255) / 256, 256, 0, stream>>>(curmax, NROW * D);
  k_feature<<<NROW * TOPK / 16, 256, 0, stream>>>(idt, w3t, b3, w4t, b4, idxb, mkv, curmax); // 5000
  k_final<<<NROW / 16 * 2, 256, 0, stream>>>(curmax, w5t, b5, tgt, seed, outTgt);
}
